// NeighborEmbedding_21208548508412
// MI455X (gfx1250) — matrix operand order repaired and checked by reading, not compiled, not run on hardware
//
#include <hip/hip_runtime.h>
#include <hip/hip_bf16.h>

typedef __bf16 bf16;
typedef __attribute__((ext_vector_type(16))) __bf16 v16bf;
typedef __attribute__((ext_vector_type(8)))  float  v8f;
typedef __attribute__((ext_vector_type(4)))  float  f32x4;

union Frag { v16bf v; f32x4 q[2]; };

#define KNN 32
#define CDIV(a,b) (((a)+(b)-1)/(b))

// ---------------------------------------------------------------------------
// 1) transpose [B,3,N] -> [B,N,3] and first 3->64 linear (K=3: plain FMA)
// ---------------------------------------------------------------------------
__global__ void emb1_kernel(const float* __restrict__ x, const float* __restrict__ w,
                            float* __restrict__ xyz, float* __restrict__ h, int B, int N) {
    int t = blockIdx.x * blockDim.x + threadIdx.x;
    if (t >= B * N) return;
    int b = t / N, n = t % N;
    float px = x[((size_t)b * 3 + 0) * N + n];
    float py = x[((size_t)b * 3 + 1) * N + n];
    float pz = x[((size_t)b * 3 + 2) * N + n];
    float* xr = xyz + (size_t)t * 3;
    xr[0] = px; xr[1] = py; xr[2] = pz;
    float* hr = h + (size_t)t * 64;
#pragma unroll 8
    for (int o = 0; o < 64; ++o)
        hr[o] = px * w[o * 3 + 0] + py * w[o * 3 + 1] + pz * w[o * 3 + 2];
}

// ---------------------------------------------------------------------------
// 2) BN batch statistics: one block per channel, deterministic tree reduce
// ---------------------------------------------------------------------------
template <typename T>
__global__ void bn_stats_kernel(const T* __restrict__ h, size_t rows, int C,
                                float* __restrict__ mean, float* __restrict__ inv) {
    int c = blockIdx.x;
    float s = 0.f, sq = 0.f;
    for (size_t r = threadIdx.x; r < rows; r += blockDim.x) {
        float v = (float)h[r * (size_t)C + c];
        s += v; sq += v * v;
    }
    __shared__ float ss[256], sqs[256];
    ss[threadIdx.x] = s; sqs[threadIdx.x] = sq;
    __syncthreads();
    for (int st = 128; st > 0; st >>= 1) {
        if (threadIdx.x < st) {
            ss[threadIdx.x]  += ss[threadIdx.x + st];
            sqs[threadIdx.x] += sqs[threadIdx.x + st];
        }
        __syncthreads();
    }
    if (threadIdx.x == 0) {
        float m = ss[0] / (float)rows;
        float var = sqs[0] / (float)rows - m * m;
        mean[c] = m;
        inv[c]  = rsqrtf(var + 1e-5f);
    }
}

// ---------------------------------------------------------------------------
// 3) BN + ReLU apply (elementwise), optional f32 / bf16 outputs
// ---------------------------------------------------------------------------
template <typename TIN>
__global__ void bn_apply_kernel(const TIN* __restrict__ h, const float* __restrict__ mean,
                                const float* __restrict__ inv, const float* __restrict__ g,
                                const float* __restrict__ bta, bf16* __restrict__ obf,
                                float* __restrict__ of32, size_t total, int C) {
    size_t t = (size_t)blockIdx.x * blockDim.x + threadIdx.x;
    if (t >= total) return;
    int c = (int)(t % C);
    float v = (float)h[t];
    float y = fmaxf((v - mean[c]) * inv[c] * g[c] + bta[c], 0.f);
    if (obf)  obf[t]  = (bf16)y;
    if (of32) of32[t] = y;
}

// ---------------------------------------------------------------------------
// 4) BN + ReLU + max over k neighbors (sg-block epilogue) -> f32 [B*S, O]
// ---------------------------------------------------------------------------
__global__ void bn_apply_max_kernel(const bf16* __restrict__ h, const float* __restrict__ mean,
                                    const float* __restrict__ inv, const float* __restrict__ g,
                                    const float* __restrict__ bta, float* __restrict__ out,
                                    size_t bs_total, int O) {
    size_t t = (size_t)blockIdx.x * blockDim.x + threadIdx.x;
    if (t >= bs_total * (size_t)O) return;
    int o = (int)(t % O);
    size_t r = t / O;
    const bf16* p = h + (r * KNN) * (size_t)O + o;
    float m = mean[o], iv = inv[o], gg = g[o], bb = bta[o];
    float best = 0.f;  // relu output is >= 0
#pragma unroll 8
    for (int j = 0; j < KNN; ++j) {
        float v = fmaxf(((float)p[(size_t)j * O] - m) * iv * gg + bb, 0.f);
        best = fmaxf(best, v);
    }
    out[t] = best;
}

// ---------------------------------------------------------------------------
// 5) WMMA GEMM:  C[M,O](bf16) = A[M,K](bf16) x W[O,K](bf16)^T + bias
//    block = 256 thr = 8 waves; 64xK weight tile staged in LDS (<=64KB);
//    wave = 16x64 tile; double-buffered A(global)/B(LDS) fragments;
//    branch-free steady-state loop + peeled tail (K is a multiple of 64);
//    v_wmma_f32_16x16x32_bf16
// ---------------------------------------------------------------------------
__global__ void __launch_bounds__(256)
gemm_bf16_wmma(const bf16* __restrict__ A, const bf16* __restrict__ W,
               const float* __restrict__ bias, bf16* __restrict__ C,
               int K, int O) {
    extern __shared__ char smem[];
    bf16* sW = (bf16*)smem;                       // [64][K], same layout as global
    const int lane = threadIdx.x & 31;
    const int wave = threadIdx.x >> 5;
    const int half = lane >> 4;
    const int l16  = lane & 15;
    const int m0 = (blockIdx.x * 8 + wave) * 16;
    const int o0 = blockIdx.y * 64;

    // cooperative stage: 64 rows x K bf16 = 8*K chunks of 16B
    {
        const f32x4* src = (const f32x4*)(W + (size_t)o0 * K);
        f32x4* dst = (f32x4*)sW;
        for (int i = threadIdx.x; i < 8 * K; i += 256) dst[i] = src[i];
    }
    __syncthreads();

    const size_t rowA = (size_t)(m0 + l16) * K;
    const bf16* sWl = sW + (size_t)l16 * K;

    auto loadA = [&](Frag& f, int k) {
        const bf16* ap = A + rowA + k + 8 * half;     // ISA 16-bit A 16x32 layout
        f.q[0] = *(const f32x4*)(ap);
        f.q[1] = *(const f32x4*)(ap + 16);
    };
    auto loadB = [&](Frag* f, int k) {
#pragma unroll
        for (int t = 0; t < 4; ++t) {                 // B 32x16 layout from LDS
            const bf16* wp = sWl + (size_t)t * 16 * K + k + 8 * half;
            f[t].q[0] = *(const f32x4*)(wp);
            f[t].q[1] = *(const f32x4*)(wp + 16);
        }
    };
    v8f acc[4] = {};
    auto mma4 = [&](Frag& a, Frag* b) {
#pragma unroll
        for (int t = 0; t < 4; ++t)
            acc[t] = __builtin_amdgcn_wmma_f32_16x16x32_bf16(
                false, a.v, false, b[t].v, (short)0, acc[t], false, false);
    };

    Frag a0, a1, b0[4], b1[4];
    loadA(a0, 0);
    loadB(b0, 0);
    int k = 0;
    // steady state: both prefetches unconditionally valid -> branch-free body
    for (; k + 64 < K; k += 64) {
        loadA(a1, k + 32); loadB(b1, k + 32);
        mma4(a0, b0);
        loadA(a0, k + 64); loadB(b0, k + 64);
        mma4(a1, b1);
    }
    // tail: k == K - 64 (K is a multiple of 64)
    loadA(a1, k + 32); loadB(b1, k + 32);
    mma4(a0, b0);
    mma4(a1, b1);

#pragma unroll
    for (int t = 0; t < 4; ++t) {
        int col = o0 + t * 16 + l16;
        float bs = bias ? bias[col] : 0.f;
#pragma unroll
        for (int r = 0; r < 8; ++r) {
            int m = 8 * half + r;
            C[(size_t)(m0 + m) * O + col] = (bf16)(acc[t][r] + bs);
        }
    }
}

// ---------------------------------------------------------------------------
// 6) Farthest point sampling: one block per batch, scan-faithful emission
// ---------------------------------------------------------------------------
__global__ void fps_kernel(const float* __restrict__ xyz, int N, int S,
                           int* __restrict__ out_idx) {
    int b = blockIdx.x;
    const float* X = xyz + (size_t)b * N * 3;
    __shared__ float dist[2048];
    __shared__ float rv[256];
    __shared__ int   ri[256];
    __shared__ int   s_far;
    __shared__ float fx, fy, fz;
    for (int i = threadIdx.x; i < N; i += blockDim.x) dist[i] = 1e10f;
    if (threadIdx.x == 0) s_far = 0;
    __syncthreads();
    for (int it = 0; it < S; ++it) {
        if (threadIdx.x == 0) {
            int far = s_far;
            out_idx[b * S + it] = far;   // emit carry (matches lax.scan)
            fx = X[far * 3 + 0]; fy = X[far * 3 + 1]; fz = X[far * 3 + 2];
        }
        __syncthreads();
        float bv = -1.f; int bi = 0;
        for (int i = threadIdx.x; i < N; i += blockDim.x) {
            float dx = X[i * 3 + 0] - fx, dy = X[i * 3 + 1] - fy, dz = X[i * 3 + 2] - fz;
            float nd = fminf(dist[i], dx * dx + dy * dy + dz * dz);
            dist[i] = nd;
            if (nd > bv) { bv = nd; bi = i; }   // ascending i => first max kept
        }
        rv[threadIdx.x] = bv; ri[threadIdx.x] = bi;
        __syncthreads();
        for (int st = 128; st > 0; st >>= 1) {
            if (threadIdx.x < st) {
                float ov = rv[threadIdx.x + st]; int oi = ri[threadIdx.x + st];
                if (ov > rv[threadIdx.x] ||
                    (ov == rv[threadIdx.x] && oi < ri[threadIdx.x])) {
                    rv[threadIdx.x] = ov; ri[threadIdx.x] = oi;
                }
            }
            __syncthreads();
        }
        if (threadIdx.x == 0) s_far = ri[0];
        __syncthreads();
    }
}

// ---------------------------------------------------------------------------
// 7) gather rows: dst[b,s,:] = src[b, idx[b,s], :]
// ---------------------------------------------------------------------------
__global__ void gather_rows_kernel(const float* __restrict__ src, const int* __restrict__ idx,
                                   float* __restrict__ dst, int N, int S, int C, size_t total) {
    size_t t = (size_t)blockIdx.x * blockDim.x + threadIdx.x;
    if (t >= total) return;
    int c = (int)(t % C);
    size_t r = t / C;
    int s = (int)(r % S);
    int b = (int)(r / S);
    dst[t] = src[((size_t)b * N + idx[b * S + s]) * C + c];
}

// ---------------------------------------------------------------------------
// 8) kNN (k=32): one block per centroid; 32 exclusion argmin passes over LDS
// ---------------------------------------------------------------------------
__global__ void knn_kernel(const float* __restrict__ coords, const float* __restrict__ cent,
                           int N, int S, int* __restrict__ knn) {
    int bs = blockIdx.x;
    int b = bs / S, s = bs % S;
    const float* X = coords + (size_t)b * N * 3;
    const float* cpt = cent + ((size_t)b * S + s) * 3;
    float cx = cpt[0], cy = cpt[1], cz = cpt[2];
    __shared__ float dist[2048];
    __shared__ float rv[256];
    __shared__ int   ri[256];
    for (int i = threadIdx.x; i < N; i += blockDim.x) {
        float dx = X[i * 3 + 0] - cx, dy = X[i * 3 + 1] - cy, dz = X[i * 3 + 2] - cz;
        dist[i] = dx * dx + dy * dy + dz * dz;
    }
    __syncthreads();
    for (int j = 0; j < KNN; ++j) {
        float bv = 3.0e38f; int bi = N;
        for (int i = threadIdx.x; i < N; i += blockDim.x) {
            float d = dist[i];
            if (d < bv) { bv = d; bi = i; }     // stable: first occurrence
        }
        rv[threadIdx.x] = bv; ri[threadIdx.x] = bi;
        __syncthreads();
        for (int st = 128; st > 0; st >>= 1) {
            if (threadIdx.x < st) {
                float ov = rv[threadIdx.x + st]; int oi = ri[threadIdx.x + st];
                if (ov < rv[threadIdx.x] ||
                    (ov == rv[threadIdx.x] && oi < ri[threadIdx.x])) {
                    rv[threadIdx.x] = ov; ri[threadIdx.x] = oi;
                }
            }
            __syncthreads();
        }
        if (threadIdx.x == 0) {
            knn[((size_t)b * S + s) * KNN + j] = ri[0];
            dist[ri[0]] = 3.4e38f;              // exclude
        }
        __syncthreads();
    }
}

// ---------------------------------------------------------------------------
// 9) build agg = [grouped - center | center] -> bf16 rows of width 2D
// ---------------------------------------------------------------------------
__global__ void agg_kernel(const float* __restrict__ feat, const int* __restrict__ fps_idx,
                           const int* __restrict__ knn, bf16* __restrict__ agg,
                           int N, int S, int D, size_t total) {
    size_t t = (size_t)blockIdx.x * blockDim.x + threadIdx.x;
    if (t >= total) return;                      // total = B*S*KNN*D
    int c = (int)(t % D);
    size_t r = t / D;
    int j = (int)(r % KNN);
    size_t ss = r / KNN;
    int s = (int)(ss % S);
    int b = (int)(ss / S);
    int ci = fps_idx[b * S + s];
    int ni = knn[((size_t)b * S + s) * KNN + j];
    float fc = feat[((size_t)b * N + ci) * D + c];
    float fn = feat[((size_t)b * N + ni) * D + c];
    bf16* row = agg + (((size_t)b * S + s) * KNN + j) * (size_t)(2 * D);
    row[c]     = (bf16)(fn - fc);
    row[D + c] = (bf16)fc;
}

// ---------------------------------------------------------------------------
// 10) misc: f32 -> bf16 convert, final [B,S,C] -> [B,C,S] transpose
// ---------------------------------------------------------------------------
__global__ void f2bf_kernel(const float* __restrict__ in, bf16* __restrict__ out, size_t n) {
    size_t t = (size_t)blockIdx.x * blockDim.x + threadIdx.x;
    if (t < n) out[t] = (bf16)in[t];
}

__global__ void transpose_out_kernel(const float* __restrict__ f, float* __restrict__ out,
                                     int S, int C, size_t total) {
    size_t t = (size_t)blockIdx.x * blockDim.x + threadIdx.x;
    if (t >= total) return;                      // t indexes f: ((b*S+s)*C+c)
    int c = (int)(t % C);
    size_t r = t / C;
    int s = (int)(r % S);
    int b = (int)(r / S);
    out[((size_t)b * C + c) * S + s] = f[t];
}

// ---------------------------------------------------------------------------
extern "C" void kernel_launch(void* const* d_in, const int* in_sizes, int n_in,
                              void* d_out, int out_size, void* d_ws, size_t ws_size,
                              hipStream_t stream) {
    const int B = 16, N = 2048, S1 = 512, S2 = 256;
    const float* x       = (const float*)d_in[0];
    const float* emb_w1  = (const float*)d_in[1];
    const float* emb_g1  = (const float*)d_in[2];
    const float* emb_b1  = (const float*)d_in[3];
    const float* emb_w2  = (const float*)d_in[4];
    const float* emb_g2  = (const float*)d_in[5];
    const float* emb_b2  = (const float*)d_in[6];
    const float* sg1_w1  = (const float*)d_in[7];
    const float* sg1_cb1 = (const float*)d_in[8];
    const float* sg1_g1  = (const float*)d_in[9];
    const float* sg1_b1  = (const float*)d_in[10];
    const float* sg1_w2  = (const float*)d_in[11];
    const float* sg1_cb2 = (const float*)d_in[12];
    const float* sg1_g2  = (const float*)d_in[13];
    const float* sg1_b2  = (const float*)d_in[14];
    const float* sg2_w1  = (const float*)d_in[15];
    const float* sg2_cb1 = (const float*)d_in[16];
    const float* sg2_g1  = (const float*)d_in[17];
    const float* sg2_b1  = (const float*)d_in[18];
    const float* sg2_w2  = (const float*)d_in[19];
    const float* sg2_cb2 = (const float*)d_in[20];
    const float* sg2_g2  = (const float*)d_in[21];
    const float* sg2_b2  = (const float*)d_in[22];

    // -------- workspace carve-out (256B aligned cursor) --------
    char* wsp = (char*)d_ws;
    size_t off = 0;
    auto alloc = [&](size_t bytes) -> void* {
        void* p = wsp + off;
        off = (off + bytes + 255) & ~(size_t)255;
        return p;
    };
    float* xyz    = (float*)alloc((size_t)B * N * 3 * 4);
    float* h_e1   = (float*)alloc((size_t)B * N * 64 * 4);
    bf16*  fe1_bf = (bf16*)alloc((size_t)B * N * 64 * 2);
    bf16*  he2_bf = (bf16*)alloc((size_t)B * N * 64 * 2);
    float* f0     = (float*)alloc((size_t)B * N * 64 * 4);
    float* mean   = (float*)alloc(512 * 4);
    float* inv    = (float*)alloc(512 * 4);
    bf16*  w_e2   = (bf16*)alloc((size_t)64 * 64 * 2);
    bf16*  w_s11  = (bf16*)alloc((size_t)128 * 128 * 2);
    bf16*  w_s12  = (bf16*)alloc((size_t)128 * 128 * 2);
    bf16*  w_s21  = (bf16*)alloc((size_t)512 * 256 * 2);
    bf16*  w_s22  = (bf16*)alloc((size_t)512 * 512 * 2);
    int*   fps1   = (int*)alloc((size_t)B * S1 * 4);
    float* xyz1   = (float*)alloc((size_t)B * S1 * 3 * 4);
    int*   knn1   = (int*)alloc((size_t)B * S1 * KNN * 4);
    float* f1     = (float*)alloc((size_t)B * S1 * 128 * 4);
    int*   fps2   = (int*)alloc((size_t)B * S2 * 4);
    float* xyz2   = (float*)alloc((size_t)B * S2 * 3 * 4);
    int*   knn2   = (int*)alloc((size_t)B * S2 * KNN * 4);
    float* f2     = (float*)alloc((size_t)B * S2 * 512 * 4);
    // big arenas with reuse (each 134,217,728 B)
    const size_t BIG = (size_t)B * S2 * KNN * 512 * 2;
    char* P0 = (char*)alloc(BIG);
    char* P1 = (char*)alloc(BIG);
    char* P2 = (char*)alloc(BIG);
    const size_t HALF = BIG / 2;                 // 67,108,864 B
    bf16* agg1 = (bf16*)P0;        bf16* h11 = (bf16*)(P0 + HALF);
    bf16* a11  = (bf16*)P1;        bf16* h12 = (bf16*)(P1 + HALF);
    bf16* agg2 = (bf16*)P0;        // reuse after stage-1 conv done
    bf16* h21  = (bf16*)P1;        // reuse after f1 extracted
    bf16* a21  = (bf16*)P2;
    bf16* h22  = (bf16*)P0;        // reuse after agg2 consumed

    const int T = 256;

    // -------- weight conversion to bf16 --------
    f2bf_kernel<<<CDIV(64 * 64, T), T, 0, stream>>>(emb_w2, w_e2, (size_t)64 * 64);
    f2bf_kernel<<<CDIV(128 * 128, T), T, 0, stream>>>(sg1_w1, w_s11, (size_t)128 * 128);
    f2bf_kernel<<<CDIV(128 * 128, T), T, 0, stream>>>(sg1_w2, w_s12, (size_t)128 * 128);
    f2bf_kernel<<<CDIV(512 * 256, T), T, 0, stream>>>(sg2_w1, w_s21, (size_t)512 * 256);
    f2bf_kernel<<<CDIV(512 * 512, T), T, 0, stream>>>(sg2_w2, w_s22, (size_t)512 * 512);

    // -------- embedding MLP --------
    size_t rowsE = (size_t)B * N;                         // 32768
    emb1_kernel<<<CDIV(B * N, T), T, 0, stream>>>(x, emb_w1, xyz, h_e1, B, N);
    bn_stats_kernel<float><<<64, T, 0, stream>>>(h_e1, rowsE, 64, mean, inv);
    bn_apply_kernel<float><<<CDIV(rowsE * 64, T), T, 0, stream>>>(
        h_e1, mean, inv, emb_g1, emb_b1, fe1_bf, nullptr, rowsE * 64, 64);
    gemm_bf16_wmma<<<dim3((unsigned)(rowsE / 128), 1), T, (size_t)64 * 64 * 2, stream>>>(
        fe1_bf, w_e2, nullptr, he2_bf, 64, 64);
    bn_stats_kernel<bf16><<<64, T, 0, stream>>>(he2_bf, rowsE, 64, mean, inv);
    bn_apply_kernel<bf16><<<CDIV(rowsE * 64, T), T, 0, stream>>>(
        he2_bf, mean, inv, emb_g2, emb_b2, nullptr, f0, rowsE * 64, 64);

    // -------- SG block 1: FPS/kNN/group + 128->128->128 (max over k) --------
    fps_kernel<<<B, T, 0, stream>>>(xyz, N, S1, fps1);
    gather_rows_kernel<<<CDIV((size_t)B * S1 * 3, T), T, 0, stream>>>(
        xyz, fps1, xyz1, N, S1, 3, (size_t)B * S1 * 3);
    knn_kernel<<<B * S1, T, 0, stream>>>(xyz, xyz1, N, S1, knn1);
    size_t aggT1 = (size_t)B * S1 * KNN * 64;
    agg_kernel<<<CDIV(aggT1, T), T, 0, stream>>>(f0, fps1, knn1, agg1, N, S1, 64, aggT1);

    size_t rows1 = (size_t)B * S1 * KNN;                  // 262144
    gemm_bf16_wmma<<<dim3((unsigned)(rows1 / 128), 2), T, (size_t)64 * 128 * 2, stream>>>(
        agg1, w_s11, sg1_cb1, h11, 128, 128);
    bn_stats_kernel<bf16><<<128, T, 0, stream>>>(h11, rows1, 128, mean, inv);
    bn_apply_kernel<bf16><<<CDIV(rows1 * 128, T), T, 0, stream>>>(
        h11, mean, inv, sg1_g1, sg1_b1, a11, nullptr, rows1 * 128, 128);
    gemm_bf16_wmma<<<dim3((unsigned)(rows1 / 128), 2), T, (size_t)64 * 128 * 2, stream>>>(
        a11, w_s12, sg1_cb2, h12, 128, 128);
    bn_stats_kernel<bf16><<<128, T, 0, stream>>>(h12, rows1, 128, mean, inv);
    bn_apply_max_kernel<<<CDIV((size_t)B * S1 * 128, T), T, 0, stream>>>(
        h12, mean, inv, sg1_g2, sg1_b2, f1, (size_t)B * S1, 128);

    // -------- SG block 2: FPS/kNN/group + 256->512->512 (max over k) --------
    fps_kernel<<<B, T, 0, stream>>>(xyz1, S1, S2, fps2);
    gather_rows_kernel<<<CDIV((size_t)B * S2 * 3, T), T, 0, stream>>>(
        xyz1, fps2, xyz2, S1, S2, 3, (size_t)B * S2 * 3);
    knn_kernel<<<B * S2, T, 0, stream>>>(xyz1, xyz2, S1, S2, knn2);
    size_t aggT2 = (size_t)B * S2 * KNN * 128;
    agg_kernel<<<CDIV(aggT2, T), T, 0, stream>>>(f1, fps2, knn2, agg2, S1, S2, 128, aggT2);

    size_t rows2 = (size_t)B * S2 * KNN;                  // 131072
    gemm_bf16_wmma<<<dim3((unsigned)(rows2 / 128), 8), T, (size_t)64 * 256 * 2, stream>>>(
        agg2, w_s21, sg2_cb1, h21, 256, 512);
    bn_stats_kernel<bf16><<<512, T, 0, stream>>>(h21, rows2, 512, mean, inv);
    bn_apply_kernel<bf16><<<CDIV(rows2 * 512, T), T, 0, stream>>>(
        h21, mean, inv, sg2_g1, sg2_b1, a21, nullptr, rows2 * 512, 512);
    gemm_bf16_wmma<<<dim3((unsigned)(rows2 / 128), 8), T, (size_t)64 * 512 * 2, stream>>>(
        a21, w_s22, sg2_cb2, h22, 512, 512);
    bn_stats_kernel<bf16><<<512, T, 0, stream>>>(h22, rows2, 512, mean, inv);
    bn_apply_max_kernel<<<CDIV((size_t)B * S2 * 512, T), T, 0, stream>>>(
        h22, mean, inv, sg2_g2, sg2_b2, f2, (size_t)B * S2, 512);

    // -------- final [B,S2,512] -> [B,512,S2] --------
    transpose_out_kernel<<<CDIV((size_t)B * S2 * 512, T), T, 0, stream>>>(
        f2, (float*)d_out, S2, 512, (size_t)B * S2 * 512);
}
